// WindowAttention_5669356833040
// MI455X (gfx1250) — compile-verified, matrix operand order checked by reading, not run on hardware
//
#include <hip/hip_runtime.h>
#include <math.h>

typedef __attribute__((ext_vector_type(16))) _Float16 v16h;
typedef __attribute__((ext_vector_type(8)))  float    v8f;

#define WP   7      // window P
#define SH   3      // shift D
#define PP   49
#define HEADS 8
#define DH   32
#define INNER 256
#define CCH  128
#define HW   56
#define N1   8
#define N2   8
#define NWIN 64
#define NB   32

// LDS row strides (padded to dodge bank conflicts)
#define QS 776    // qkv f16 rows (768 + 8)
#define XS 136    // x f16 rows (128 + 8)
#define MS 52     // mask f32 rows
#define SS 56     // scores f32 rows
#define AS 72     // attn f16 rows (64 + 8)
#define OS 264    // o f16 rows (256 + 8)

#define OFF_QKV  0
#define OFF_X    (OFF_QKV + 64*QS*2)   // aliased: sX (phase 0/1), sP attn f32 (phase 2)
#define OFF_MASK (OFF_X   + 64*XS*2)
#define OFF_S    (OFF_MASK + PP*MS*4)
#define OFF_ATT  (OFF_S   + 64*SS*4)
#define OFF_O    (OFF_ATT + 64*AS*2)
#define OFF_RMAX (OFF_O   + 64*OS*2)
#define OFF_RINV (OFF_RMAX + 64*4)
#define LDS_BYTES (OFF_RINV + 64*4)

// ---- WMMA fragment loaders (wave32, 16x16x32 f16) -------------------------
// A 16x32: lanes 0-15 hold row M=lane with K = {0..7,16..23}; lanes 16-31 hold
// row M=lane-16 with K = {8..15,24..31}.
__device__ inline v16h frag_rows(const _Float16* __restrict__ base, int stride) {
  const int lane = threadIdx.x & 31;
  const _Float16* p = base + (lane & 15) * stride + ((lane >> 4) << 3);
  v16h f;
#pragma unroll
  for (int t = 0; t < 8; ++t) { f[t] = p[t]; f[t + 8] = p[16 + t]; }
  return f;
}
// B 32x16 loaded as A-layout of B^T: lane holds column N = lane&15.
__device__ inline v16h frag_cols(const _Float16* __restrict__ base, int stride) {
  const int lane = threadIdx.x & 31;
  const int col  = lane & 15;
  const int kb   = (lane >> 4) << 3;
  v16h f;
#pragma unroll
  for (int t = 0; t < 8; ++t) {
    f[t]     = base[(kb + t) * stride + col];
    f[t + 8] = base[(kb + 16 + t) * stride + col];
  }
  return f;
}

// Async LDS -> global store (CDNA5, tracked on ASYNCcnt).
__device__ inline void async_store_b32(const float* gptr, uint32_t lds_byte_off) {
  uint64_t ga = (uint64_t)gptr;
  asm volatile("global_store_async_from_lds_b32 %0, %1, off"
               :: "v"(ga), "v"(lds_byte_off) : "memory");
}

__global__ void swin_cvt_weights(const float* __restrict__ wqkv,
                                 const float* __restrict__ wout,
                                 _Float16* __restrict__ wq16,
                                 _Float16* __restrict__ wo16) {
  int i = blockIdx.x * blockDim.x + threadIdx.x;
  if (i < 3 * INNER * CCH) wq16[i] = (_Float16)wqkv[i];
  if (i < CCH * INNER)     wo16[i] = (_Float16)wout[i];
}

__global__ __launch_bounds__(256) void swin_window_attn(
    const float* __restrict__ x,
    const _Float16* __restrict__ wqkv,
    const _Float16* __restrict__ wout,
    const float* __restrict__ bout,
    const float* __restrict__ pos,
    float* __restrict__ out_o,
    float* __restrict__ out_attn) {
  extern __shared__ char smem[];
  _Float16* sQKV = (_Float16*)(smem + OFF_QKV);
  _Float16* sX   = (_Float16*)(smem + OFF_X);
  float*    sP   = (float*)   (smem + OFF_X);    // attn f32 staging (aliases sX)
  float*    sM   = (float*)   (smem + OFF_MASK);
  float*    sS   = (float*)   (smem + OFF_S);
  _Float16* sA   = (_Float16*)(smem + OFF_ATT);
  _Float16* sO   = (_Float16*)(smem + OFF_O);
  float*    sMax = (float*)   (smem + OFF_RMAX);
  float*    sInv = (float*)   (smem + OFF_RINV);

  const int tid  = threadIdx.x;
  const int wv   = tid >> 5;          // wave id (8 waves)
  const int lane = tid & 31;
  const int lo   = lane & 15;
  const int hi8  = (lane >> 4) * 8;   // C/D fragment M offset

  const int wg   = blockIdx.x;        // window index, 0..2047
  const int b    = wg / NWIN;
  const int wloc = wg % NWIN;
  const int wy   = wloc / N2;
  const int wx   = wloc % N2;

  // warm L2 for phase-3 weights (64KB of w_out)
  if (tid < 256) __builtin_prefetch((const char*)wout + tid * 256, 0, 0);

  // ---------------- phase 0: load rolled window + build mask ----------------
  for (int idx = tid; idx < CCH * 64; idx += 256) {
    int c = idx >> 6, p = idx & 63;
    _Float16 v = (_Float16)0.f;
    if (p < PP) {
      int py = p / WP, px = p % WP;
      int gh = (wy * WP + py + SH) % HW;   // roll(-SH) fused
      int gw = (wx * WP + px + SH) % HW;
      v = (_Float16)x[(((size_t)b * CCH + c) * HW + gh) * HW + gw];
    }
    sX[p * XS + c] = v;                    // rows 49..63 zero
  }
  const bool ulw = (wy == N1 - 1);
  const int  lrStart = N1 * N2 - N1 - 1;   // faithful to lr[-n1-1::n1]
  const bool lrw = (wloc >= lrStart) && (((wloc - lrStart) % N1) == 0);
  for (int idx = tid; idx < PP * PP; idx += 256) {
    int i = idx / PP, j = idx % PP;
    int dx = (j / WP) - (i / WP) + (WP - 1);
    int dy = (j % WP) - (i % WP) + (WP - 1);
    float m = pos[dx * (2 * WP - 1) + dy];
    if (ulw && ((i >= PP - SH * WP) != (j >= PP - SH * WP))) m = -1e30f;
    if (lrw && (((i % WP) >= WP - SH) != ((j % WP) >= WP - SH))) m = -1e30f;
    sM[i * MS + j] = m;
  }
  __syncthreads();

  // ---------------- phase 1: QKV = Xw @ Wqkv^T (49x128 @ 128x768) ----------
  const float scale = 0.17677669529663687f;   // 32^-0.5, folded into Q
  for (int t = wv; t < 4 * 48; t += 8) {
    int mt = t / 48, nt = t % 48;
    v8f acc = {};
#pragma unroll
    for (int ks = 0; ks < 4; ++ks) {
      v16h a  = frag_rows(sX + mt * 16 * XS + ks * 32, XS);
      v16h bf = frag_rows(wqkv + (size_t)(nt * 16) * CCH + ks * 32, CCH);
      acc = __builtin_amdgcn_wmma_f32_16x16x32_f16(false, a, false, bf,
                                                   (short)0, acc, false, false);
    }
    int col = nt * 16 + lo;
    float s = (col < INNER) ? scale : 1.f;
#pragma unroll
    for (int r = 0; r < 8; ++r)
      sQKV[(mt * 16 + hi8 + r) * QS + col] = (_Float16)(acc[r] * s);
  }
  __syncthreads();   // also: sX dead from here; its LDS is reused as sP

  // ---------------- phase 2: per-head attention -----------------------------
  for (int h = 0; h < HEADS; ++h) {
    // S = Q K^T + mask   (16 tiles of 16x16, K=32)
    for (int t = wv; t < 16; t += 8) {
      int mt = t >> 2, nt = t & 3;
      v16h a  = frag_rows(sQKV + mt * 16 * QS + h * DH, QS);
      v16h bf = frag_rows(sQKV + nt * 16 * QS + INNER + h * DH, QS);
      v8f acc = {};
      acc = __builtin_amdgcn_wmma_f32_16x16x32_f16(false, a, false, bf,
                                                   (short)0, acc, false, false);
      int j = nt * 16 + lo;
      if (j < PP) {
#pragma unroll
        for (int r = 0; r < 8; ++r) {
          int i = mt * 16 + hi8 + r;
          float v = acc[r];
          if (i < PP) v += sM[i * MS + j];
          sS[i * SS + j] = v;
        }
      }
    }
    __syncthreads();

    // row softmax stats over 49 valid columns
    if (tid < PP) {
      const float* row = sS + tid * SS;
      float mx = -1e30f;
      for (int j = 0; j < PP; ++j) mx = fmaxf(mx, row[j]);
      float sum = 0.f;
      for (int j = 0; j < PP; ++j) sum += __expf(row[j] - mx);
      sMax[tid] = mx;
      sInv[tid] = 1.f / sum;
    }
    __syncthreads();

    // previous head's async drain of sP must be complete before refilling:
    // a thread's overwrite set (i*PP+j) is not its own async-store set (idx).
    asm volatile("s_wait_asynccnt 0x0" ::: "memory");

    // normalize: f32 tile to sP (packed 49x49), f16 zero-padded 64x64 to sA
    for (int idx = tid; idx < 64 * 64; idx += 256) {
      int i = idx >> 6, j = idx & 63;
      float p = 0.f;
      if (i < PP && j < PP) {
        p = __expf(sS[i * SS + j] - sMax[i]) * sInv[i];
        sP[i * PP + j] = p;
      }
      sA[i * AS + j] = (_Float16)p;
    }

    // kick off async DMA of the attn tile (LDS -> global) and keep computing
    asm volatile("s_wait_dscnt 0x0" ::: "memory");
    {
      float* gbase = out_attn + (((size_t)wg * HEADS + h) * PP) * PP;
      for (int idx = tid; idx < PP * PP; idx += 256)
        async_store_b32(gbase + idx, (uint32_t)(uintptr_t)(sP + idx));
    }
    __syncthreads();

    // O_head = attn @ V   (8 tiles: 4 M-tiles x 2 N-tiles, K=64)
    {
      int mt = wv >> 1, nt = wv & 1;   // exactly one tile per wave
      v8f acc = {};
#pragma unroll
      for (int ks = 0; ks < 2; ++ks) {
        v16h a  = frag_rows(sA + mt * 16 * AS + ks * 32, AS);
        v16h bf = frag_cols(sQKV + (ks * 32) * QS + 2 * INNER + h * DH + nt * 16, QS);
        acc = __builtin_amdgcn_wmma_f32_16x16x32_f16(false, a, false, bf,
                                                     (short)0, acc, false, false);
      }
      int col = h * DH + nt * 16 + lo;
#pragma unroll
      for (int r = 0; r < 8; ++r)
        sO[(mt * 16 + hi8 + r) * OS + col] = (_Float16)acc[r];
    }
    __syncthreads();
  }

  // ---------------- phase 3: out = O @ Wout^T + b, scatter with roll(+SH) ---
  for (int t = wv; t < 32; t += 8) {
    int mt = t >> 3, nt = t & 7;
    v8f acc = {};
#pragma unroll
    for (int ks = 0; ks < 8; ++ks) {
      v16h a  = frag_rows(sO + mt * 16 * OS + ks * 32, OS);
      v16h bf = frag_rows(wout + (size_t)(nt * 16) * INNER + ks * 32, INNER);
      acc = __builtin_amdgcn_wmma_f32_16x16x32_f16(false, a, false, bf,
                                                   (short)0, acc, false, false);
    }
    int c = nt * 16 + lo;
    float bb = bout[c];
#pragma unroll
    for (int r = 0; r < 8; ++r) {
      int i = mt * 16 + hi8 + r;
      if (i < PP) {
        int py = i / WP, px = i % WP;
        int gh = (wy * WP + py + SH) % HW;   // roll(+SH) fused
        int gw = (wx * WP + px + SH) % HW;
        out_o[(((size_t)b * CCH + c) * HW + gh) * HW + gw] = acc[r] + bb;
      }
    }
  }
  // s_endpgm performs an implicit wait-idle, draining remaining ASYNCcnt.
}

extern "C" void kernel_launch(void* const* d_in, const int* in_sizes, int n_in,
                              void* d_out, int out_size, void* d_ws, size_t ws_size,
                              hipStream_t stream) {
  (void)in_sizes; (void)n_in; (void)out_size; (void)ws_size;
  const float* x     = (const float*)d_in[0];
  const float* wqkv  = (const float*)d_in[1];
  const float* wout  = (const float*)d_in[2];
  const float* bout  = (const float*)d_in[3];
  const float* pos   = (const float*)d_in[4];

  _Float16* wq16 = (_Float16*)d_ws;                 // 768*128 f16
  _Float16* wo16 = wq16 + (size_t)3 * INNER * CCH;  // 128*256 f16

  float* out_o    = (float*)d_out;
  float* out_attn = out_o + (size_t)NB * CCH * HW * HW;

  swin_cvt_weights<<<(3 * INNER * CCH + 255) / 256, 256, 0, stream>>>(
      wqkv, wout, wq16, wo16);

  static_assert(LDS_BYTES < 320 * 1024, "LDS budget");
  hipFuncSetAttribute((const void*)swin_window_attn,
                      hipFuncAttributeMaxDynamicSharedMemorySize, LDS_BYTES);
  swin_window_attn<<<NB * NWIN, 256, LDS_BYTES, stream>>>(
      x, wq16, wo16, bout, pos, out_o, out_attn);
}
